// FingerStateMaskGenerator_601295421861
// MI455X (gfx1250) — matrix operand order repaired and checked by reading, not compiled, not run on hardware
//
#include <hip/hip_runtime.h>

#define T_LEN   100000
#define NROWS   128         // B*2 = 64*2
#define NCH     9
#define LPAD    5
#define RPADP1  6           // RPAD + 1
#define THREADS 256
#define ITEMS   16
#define CHUNK   (THREADS * ITEMS)                   // 4096
#define NCHUNKS ((T_LEN + CHUNK - 1) / CHUNK)       // 25
#define NBLK    (NROWS * NCHUNKS)                   // 3200

typedef __attribute__((ext_vector_type(16))) _Float16 v16h;
typedef __attribute__((ext_vector_type(8)))  float    v8f;

__device__ __forceinline__ int imin(int a, int b) { return a < b ? a : b; }

// Load 16 contiguous floats; full spans use float4 (b128), tail is guarded.
__device__ __forceinline__ void load16(const float* __restrict__ src, int base,
                                       float v[ITEMS]) {
  if (base + ITEMS <= T_LEN) {
    const float4* s4 = (const float4*)(src + base);   // 64B-aligned
    #pragma unroll
    for (int q = 0; q < 4; ++q) {
      float4 x = s4[q];
      v[4*q+0] = x.x; v[4*q+1] = x.y; v[4*q+2] = x.z; v[4*q+3] = x.w;
    }
  } else {
    #pragma unroll
    for (int i = 0; i < ITEMS; ++i) {
      int t = base + i;
      v[i] = (t < T_LEN) ? src[t] : 0.0f;
    }
  }
}

// ---------------------------------------------------------------------------
// Kernel 0: zero the difference array (d_out reused as d-array) and flags.
// ---------------------------------------------------------------------------
__global__ void zero_kernel(float* __restrict__ d, int* __restrict__ flags, long n4) {
  long i = (long)blockIdx.x * blockDim.x + threadIdx.x;
  long stride = (long)gridDim.x * blockDim.x;
  float4* d4 = (float4*)d;
  float4 z = make_float4(0.f, 0.f, 0.f, 0.f);
  for (long j = i; j < n4; j += stride) d4[j] = z;
  if (i < 2 * NROWS) flags[i] = 0;
}

// ---------------------------------------------------------------------------
// Kernel 1: per-chunk minimum release-onset index (3200 independent blocks).
// ---------------------------------------------------------------------------
__global__ void relmin_kernel(const float* __restrict__ labels,
                              int* __restrict__ blockRelMin) {
  const int blk = blockIdx.x;
  const int row = blk / NCHUNKS, c = blk % NCHUNKS;
  const int b = row >> 1, f = row & 1;
  const float* rel = labels + ((long)b * NCH + (f ? 3 : 1)) * T_LEN;
  const int tid = threadIdx.x;
  const int base = c * CHUNK + tid * ITEMS;

  float rv[ITEMS];
  load16(rel, base, rv);
  float prev = (base > 0 && base <= T_LEN) ? rel[base - 1] : 0.0f;
  int tmin = T_LEN;
  #pragma unroll
  for (int i = 0; i < ITEMS; ++i) {
    int t = base + i;
    if (t < T_LEN && rv[i] > prev) tmin = imin(tmin, t);
    prev = rv[i];
  }

  __shared__ int smin[THREADS];
  smin[tid] = tmin;
  __syncthreads();
  for (int off = THREADS / 2; off >= 1; off >>= 1) {
    if (tid < off) smin[tid] = imin(smin[tid], smin[tid + off]);
    __syncthreads();
  }
  if (tid == 0) blockRelMin[blk] = smin[0];
}

// ---------------------------------------------------------------------------
// Kernel 2: per-row suffix scan of the 25 chunk minima (tiny) + anyR flag.
// ---------------------------------------------------------------------------
__global__ void relscan_kernel(const int* __restrict__ blockRelMin,
                               int* __restrict__ relCarry,
                               int* __restrict__ flags) {
  const int r = threadIdx.x;
  if (r >= NROWS) return;
  int m = T_LEN;
  for (int c = NCHUNKS - 1; c >= 0; --c) {
    relCarry[r * NCHUNKS + c] = m;                 // min over chunks strictly after c
    m = imin(m, blockRelMin[r * NCHUNKS + c]);
  }
  flags[2 * r + 1] = (m < T_LEN) ? 1 : 0;          // any release onset
}

// ---------------------------------------------------------------------------
// Kernel 3: fully parallel scatter. Intra-chunk suffix-min + precomputed
// inter-chunk carry -> next_rel per element; press onsets scatter +1/-1.
// ---------------------------------------------------------------------------
__global__ void scatter_kernel(const float* __restrict__ labels,
                               float* __restrict__ darr,
                               const int* __restrict__ relCarry,
                               int* __restrict__ flags) {
  const int blk = blockIdx.x;
  const int row = blk / NCHUNKS, c = blk % NCHUNKS;
  const int b = row >> 1, f = row & 1;
  const float* press = labels + ((long)b * NCH + (f ? 2 : 0)) * T_LEN;
  const float* rel   = press + T_LEN;
  float* drow = darr + (long)row * T_LEN;
  const int tid = threadIdx.x;
  const int base = c * CHUNK + tid * ITEMS;

  // release onsets -> relIdx
  float rv[ITEMS];
  load16(rel, base, rv);
  float prev = (base > 0 && base <= T_LEN) ? rel[base - 1] : 0.0f;
  int relIdx[ITEMS];
  int tmin = T_LEN;
  #pragma unroll
  for (int i = 0; i < ITEMS; ++i) {
    int t = base + i;
    bool on = (t < T_LEN) && (rv[i] > prev);
    relIdx[i] = on ? t : T_LEN;
    tmin = imin(tmin, relIdx[i]);
    prev = rv[i];
  }

  // intra-chunk suffix-min scan of per-thread minima
  __shared__ int smin[THREADS];
  smin[tid] = tmin;
  __syncthreads();
  for (int off = 1; off < THREADS; off <<= 1) {
    int v = smin[tid];
    if (tid + off < THREADS) v = imin(v, smin[tid + off]);
    __syncthreads();
    smin[tid] = v;
    __syncthreads();
  }
  const int carry = relCarry[blk];
  int run = imin((tid + 1 < THREADS) ? smin[tid + 1] : T_LEN, carry);

  int nextRel[ITEMS];
  #pragma unroll
  for (int i = ITEMS - 1; i >= 0; --i) {
    nextRel[i] = run;                    // min release-onset index s > t
    run = imin(run, relIdx[i]);
  }

  // press onsets -> scatter difference-array endpoints
  float pv[ITEMS];
  load16(press, base, pv);
  float pprev = (base > 0 && base <= T_LEN) ? press[base - 1] : 0.0f;
  bool localP = false;
  #pragma unroll
  for (int i = 0; i < ITEMS; ++i) {
    int t = base + i;
    bool on = (t < T_LEN) && (pv[i] > pprev);
    pprev = pv[i];
    if (on) {
      localP = true;
      int r2 = nextRel[i];
      if (r2 >= T_LEN) r2 = T_LEN - 1;           // fallback: last step
      int s = t - LPAD; if (s < 0) s = 0;
      int e = r2 + RPADP1;                       // end (exclusive)
      atomicAdd(&drow[s], 1.0f);
      if (e < T_LEN) atomicAdd(&drow[e], -1.0f); // e>=T slot dropped by ref
    }
  }

  __shared__ int sAnyP;
  if (tid == 0) sAnyP = 0;
  __syncthreads();
  if (localP) sAnyP = 1;
  __syncthreads();
  if (tid == 0 && sAnyP) atomicOr(&flags[2 * row], 1);
}

// ---------------------------------------------------------------------------
// Kernel 4: per-chunk sums of the difference array (3200 independent blocks).
// ---------------------------------------------------------------------------
__global__ void sum_kernel(const float* __restrict__ darr,
                           float* __restrict__ blockSum) {
  const int blk = blockIdx.x;
  const int row = blk / NCHUNKS, c = blk % NCHUNKS;
  const float* drow = darr + (long)row * T_LEN;
  const int tid = threadIdx.x;
  const int base = c * CHUNK + tid * ITEMS;

  float v[ITEMS];
  load16(drow, base, v);
  float s = 0.0f;
  #pragma unroll
  for (int i = 0; i < ITEMS; ++i) s += v[i];

  __shared__ float ssum[THREADS];
  ssum[tid] = s;
  __syncthreads();
  for (int off = THREADS / 2; off >= 1; off >>= 1) {
    if (tid < off) ssum[tid] += ssum[tid + off];
    __syncthreads();
  }
  if (tid == 0) blockSum[blk] = ssum[0];
}

// ---------------------------------------------------------------------------
// Kernel 5: per-row exclusive prefix of the 25 chunk sums (tiny, exact ints).
// ---------------------------------------------------------------------------
__global__ void sumscan_kernel(const float* __restrict__ blockSum,
                               float* __restrict__ sumCarry) {
  const int r = threadIdx.x;
  if (r >= NROWS) return;
  float s = 0.0f;
  for (int c = 0; c < NCHUNKS; ++c) {
    sumCarry[r * NCHUNKS + c] = s;
    s += blockSum[r * NCHUNKS + c];
  }
}

// ---------------------------------------------------------------------------
// Kernel 6: fully parallel mask apply. WMMA tensor-core block scan: the 256
// thread sums form a 16x16 A-tile (K padded to 32 with zeros) multiplied by
// an upper-triangular ones B with v_wmma_f32_16x16x32_f16 -> all intra-row
// inclusive prefixes in one matrix op (small exact ints; f32 accumulate).
// ---------------------------------------------------------------------------
__global__ void mask_kernel(float* __restrict__ darr,
                            const float* __restrict__ sumCarry,
                            const int* __restrict__ flags) {
  const int blk = blockIdx.x;
  const int row = blk / NCHUNKS, c = blk % NCHUNKS;
  float* drow = darr + (long)row * T_LEN;
  const int tid = threadIdx.x;
  const int base = c * CHUNK + tid * ITEMS;
  const bool valid = (flags[2 * row] != 0) && (flags[2 * row + 1] != 0);

  // per-thread inclusive prefix of 16 contiguous d values
  float v[ITEMS], p[ITEMS];
  load16(drow, base, v);
  float acc = 0.0f;
  #pragma unroll
  for (int i = 0; i < ITEMS; ++i) { acc += v[i]; p[i] = acc; }

  __shared__ float sums[THREADS];
  __shared__ float scanIncl[THREADS];
  __shared__ float rowExcl[16];
  sums[tid] = acc;
  __syncthreads();

  // WMMA block scan of the 256 thread sums (wave 0, EXEC all ones)
  if (tid < 32) {
    const int half = tid >> 4;   // A f16 16x32 layout: lanes 0-15 carry K0..7
    const int m    = tid & 15;   // (elems 0-7) / K16..23 (elems 8-15, zeroed);
                                 // lanes 16-31 carry K8..15 / K24..31 of row m.
    v16h a, bmat;
    #pragma unroll
    for (int e = 0; e < 16; ++e) {
      float av = (e < 8) ? sums[m * 16 + half * 8 + e] : 0.0f;
      a[e] = (_Float16)av;
      // B 32x16: lane n<16 holds B[K=0..7][n] (elems 0-7) / K16..23 (zero); lanes>=16 hold K8..15 / K24..31 (zero)
      float bv = (e < 8 && half * 8 + e <= m) ? 1.0f : 0.0f;   // B[k][n] = (k<=n)
      bmat[e] = (_Float16)bv;
    }
    v8f cz = {};
    v8f d1 = __builtin_amdgcn_wmma_f32_16x16x32_f16(
        /*neg_a=*/false, a, /*neg_b=*/false, bmat,
        /*c_mod=*/(short)0, cz, /*reuse_a=*/false, /*reuse_b=*/false);
    // D layout: VGPR q, lane n<16 -> [M=q][n]; lane n+16 -> [M=q+8][n]
    #pragma unroll
    for (int q = 0; q < 8; ++q)
      scanIncl[(q + 8 * half) * 16 + m] = d1[q];
  }
  __syncthreads();

  if (tid == 0) {
    float r = 0.0f;
    rowExcl[0] = 0.0f;
    for (int mm = 1; mm < 16; ++mm) {
      r += scanIncl[(mm - 1) * 16 + 15];   // row totals
      rowExcl[mm] = r;
    }
  }
  __syncthreads();

  const float excl = sumCarry[blk] + rowExcl[tid >> 4] + scanIncl[tid] - acc;

  float m[ITEMS];
  #pragma unroll
  for (int i = 0; i < ITEMS; ++i)
    m[i] = (valid && (excl + p[i] > 0.5f)) ? 1.0f : 0.0f;

  if (base + ITEMS <= T_LEN) {
    float4* d4 = (float4*)(drow + base);
    #pragma unroll
    for (int q = 0; q < 4; ++q)
      d4[q] = make_float4(m[4*q+0], m[4*q+1], m[4*q+2], m[4*q+3]);
  } else {
    #pragma unroll
    for (int i = 0; i < ITEMS; ++i) {
      int t = base + i;
      if (t < T_LEN) drow[t] = m[i];
    }
  }
}

// ---------------------------------------------------------------------------
extern "C" void kernel_launch(void* const* d_in, const int* in_sizes, int n_in,
                              void* d_out, int out_size, void* d_ws, size_t ws_size,
                              hipStream_t stream) {
  const float* labels = (const float*)d_in[0];   // [64, 9, 100000] float32
  float* out = (float*)d_out;                    // [64, 2, 100000] float32

  int*   flags       = (int*)d_ws;               // 2*NROWS ints
  int*   blockRelMin = flags + 2 * NROWS;        // NBLK ints
  int*   relCarry    = blockRelMin + NBLK;       // NBLK ints
  float* blockSum    = (float*)(relCarry + NBLK);// NBLK floats
  float* sumCarry    = blockSum + NBLK;          // NBLK floats

  const long n4 = (long)NROWS * T_LEN / 4;
  zero_kernel<<<2048, 256, 0, stream>>>(out, flags, n4);
  relmin_kernel<<<NBLK, THREADS, 0, stream>>>(labels, blockRelMin);
  relscan_kernel<<<1, 128, 0, stream>>>(blockRelMin, relCarry, flags);
  scatter_kernel<<<NBLK, THREADS, 0, stream>>>(labels, out, relCarry, flags);
  sum_kernel<<<NBLK, THREADS, 0, stream>>>(out, blockSum);
  sumscan_kernel<<<1, 128, 0, stream>>>(blockSum, sumCarry);
  mask_kernel<<<NBLK, THREADS, 0, stream>>>(out, sumCarry, flags);
}
